// Explainer_30039001268380
// MI455X (gfx1250) — matrix operand order repaired and checked by reading, not compiled, not run on hardware
//
#include <hip/hip_runtime.h>
#include <hip/hip_bf16.h>
#include <stdint.h>

typedef __attribute__((ext_vector_type(16))) __bf16 v16bf;
typedef __attribute__((ext_vector_type(8)))  __bf16 v8bf;
typedef __attribute__((ext_vector_type(4)))  __bf16 v4bf;
typedef __attribute__((ext_vector_type(2)))  __bf16 v2bf;
typedef __attribute__((ext_vector_type(8)))  float  v8f;

// packed f32x2 -> bf16x2 (hardware v_cvt_pk_bf16_f32 when available)
static __device__ inline v2bf pk2(float x, float y) {
#if __has_builtin(__builtin_amdgcn_cvt_pk_bf16_f32)
    return __builtin_amdgcn_cvt_pk_bf16_f32(x, y);
#else
    v2bf r; r[0] = (__bf16)x; r[1] = (__bf16)y; return r;
#endif
}

static __device__ inline v16bf cvt_row(const float4& a0, const float4& a1,
                                       const float4& a2, const float4& a3) {
    v2bf p0 = pk2(a0.x, a0.y), p1 = pk2(a0.z, a0.w);
    v2bf p2 = pk2(a1.x, a1.y), p3 = pk2(a1.z, a1.w);
    v2bf p4 = pk2(a2.x, a2.y), p5 = pk2(a2.z, a2.w);
    v2bf p6 = pk2(a3.x, a3.y), p7 = pk2(a3.z, a3.w);
    v4bf q0 = __builtin_shufflevector(p0, p1, 0,1,2,3);
    v4bf q1 = __builtin_shufflevector(p2, p3, 0,1,2,3);
    v4bf q2 = __builtin_shufflevector(p4, p5, 0,1,2,3);
    v4bf q3 = __builtin_shufflevector(p6, p7, 0,1,2,3);
    v8bf r0 = __builtin_shufflevector(q0, q1, 0,1,2,3,4,5,6,7);
    v8bf r1 = __builtin_shufflevector(q2, q3, 0,1,2,3,4,5,6,7);
    return __builtin_shufflevector(r0, r1, 0,1,2,3,4,5,6,7,8,9,10,11,12,13,14,15);
}

// ---------------------------------------------------------------------------
// WMMA GEMM: C[M x 128] = A[M x K] @ W[K x 128] (+ bias), A row-major (LDA),
// W row-major (ld 128), C row-major (ld 128). K in {128, 384} (compile-time).
// Block = 256 threads = 8 waves; block tile = 128 rows x 16 cols;
// each wave computes one 16x16 tile via v_wmma_f32_16x16x32_bf16.
// ---------------------------------------------------------------------------
template<int K, int LDA>
__global__ __launch_bounds__(256)
void k_gemm_wmma(const float* __restrict__ A, const float* __restrict__ W,
                 const float* __restrict__ bias, float* __restrict__ C, int M)
{
    constexpr int Kpad = K + 8;   // 16B-aligned per-column stride, conflict-free
    __shared__ __align__(16) __bf16 sW[16 * Kpad];
    const int colbase  = blockIdx.y * 16;
    const int rowblock = blockIdx.x * 128;
    const int t = threadIdx.x;

    // Stage W[:, colbase:colbase+16] into LDS as bf16, column-major: sW[n*Kpad + k]
    // (two k per thread -> packed v2bf -> single b32 LDS store)
    for (int i = t; i < (K / 2) * 16; i += 256) {
        int k = (i >> 4) * 2, n = i & 15;
        v2bf p = pk2(W[k * 128 + colbase + n], W[(k + 1) * 128 + colbase + n]);
        *(v2bf*)&sW[n * Kpad + k] = p;
    }
    __syncthreads();

    const int wave = t >> 5, lane = t & 31;
    const int n    = lane & 15;          // column within tile (B/C/D layout)
    const int half = lane >> 4;          // half-wave selects K-group
    const int m0   = rowblock + wave * 16;
    const int mrow = m0 + (lane & 15);   // A-row (A layout: M = lane%16)
    const int mload = (mrow < M) ? mrow : (M - 1);   // clamp; store is guarded
    const float* arow = A + (size_t)mload * LDA;

    v8f acc = {};
    #pragma unroll
    for (int kb = 0; kb < K; kb += 32) {
        // ---- A fragment: 16x32 bf16, lane holds K=[g0..g0+7] and [g1..g1+7]
        const int g0 = kb + half * 8;
        const int g1 = kb + 16 + half * 8;
        float4 a0 = *(const float4*)(arow + g0);
        float4 a1 = *(const float4*)(arow + g0 + 4);
        float4 a2 = *(const float4*)(arow + g1);
        float4 a3 = *(const float4*)(arow + g1 + 4);
        v16bf af = cvt_row(a0, a1, a2, a3);

        // ---- B fragment: 32x16 bf16, lane holds K=[g0..g0+7] and [g1..g1+7] at fixed column n
        const int bbase = n * Kpad + kb + half * 8;      // 16B-aligned by construction
        v8bf b0 = *(const v8bf*)&sW[bbase];
        v8bf b1 = *(const v8bf*)&sW[bbase + 16];
        v16bf bfrag = __builtin_shufflevector(b0, b1,
                          0,1,2,3,4,5,6,7,8,9,10,11,12,13,14,15);

        acc = __builtin_amdgcn_wmma_f32_16x16x32_bf16(
                  false, af, false, bfrag, (short)0, acc, false, false);
    }

    const float bv = bias ? bias[colbase + n] : 0.0f;
    // M is a multiple of 16 -> a wave tile is fully in-range or fully out.
    if (m0 + 15 < M) {
        float* cp = C + (size_t)(m0 + half * 8) * 128 + colbase + n;
        #pragma unroll
        for (int v = 0; v < 8; ++v) cp[v * 128] = acc[v] + bv;   // imm-offset stores
    } else {
        #pragma unroll
        for (int v = 0; v < 8; ++v) {
            int m = m0 + half * 8 + v;
            if (m < M) C[(size_t)m * 128 + colbase + n] = acc[v] + bv;
        }
    }
}

// ---------------------------------------------------------------------------
// Graph kernels
// ---------------------------------------------------------------------------
__global__ void k_deg_init(float* __restrict__ deg, int N) {
    int i = blockIdx.x * blockDim.x + threadIdx.x;
    if (i < N) deg[i] = 1.0f;                 // self-loop
}

__global__ void k_deg_acc(const int* __restrict__ dst, float* __restrict__ deg, int E) {
    int e = blockIdx.x * blockDim.x + threadIdx.x;
    if (e < E) atomicAdd(&deg[dst[e]], 1.0f);
}

__global__ void k_dinv(float* __restrict__ deg, int N) {
    int i = blockIdx.x * blockDim.x + threadIdx.x;
    if (i < N) deg[i] = rsqrtf(deg[i]);       // deg >= 1 always (self-loops)
}

// out = xcat + layer*128, row stride 384: write self-loop term (non-atomic init)
__global__ void k_selfloop(const float* __restrict__ h, const float* __restrict__ dinv,
                           float* __restrict__ out, int N) {
    int gid = blockIdx.x * blockDim.x + threadIdx.x;      // N*32 threads
    int i = gid >> 5;
    if (i >= N) return;
    int c = (gid & 31) * 4;
    float w = dinv[i]; w *= w;
    float4 hv = *(const float4*)(h + (size_t)i * 128 + c);
    float4 o; o.x = hv.x * w; o.y = hv.y * w; o.z = hv.z * w; o.w = hv.w * w;
    *(float4*)(out + (size_t)i * 384 + c) = o;
}

// one 32-lane wave per edge; 4 channels per lane (float4 gather, 4 f32 atomics)
__global__ void k_edge_scatter(const float* __restrict__ h, const float* __restrict__ dinv,
                               const int* __restrict__ src, const int* __restrict__ dst,
                               float* __restrict__ out, int E) {
    int gid = blockIdx.x * blockDim.x + threadIdx.x;      // E*32 threads
    int e = gid >> 5;
    if (e >= E) return;
    int lane = gid & 31;
    int s = src[e], d = dst[e];
    float w = dinv[s] * dinv[d];
    int c = lane * 4;
    float4 hv = *(const float4*)(h + (size_t)s * 128 + c);
    float* o = out + (size_t)d * 384 + c;
    atomicAdd(o + 0, hv.x * w);
    atomicAdd(o + 1, hv.y * w);
    atomicAdd(o + 2, hv.z * w);
    atomicAdd(o + 3, hv.w * w);
}

// in-place x = relu(x + b) on an xcat slice (row stride 384)
__global__ void k_bias_relu(float* __restrict__ x, const float* __restrict__ b, int N) {
    int gid = blockIdx.x * blockDim.x + threadIdx.x;      // N*32 threads
    int i = gid >> 5;
    if (i >= N) return;
    int c = (gid & 31) * 4;
    float* p = x + (size_t)i * 384 + c;
    float4 v = *(float4*)p;
    v.x = fmaxf(v.x + b[c + 0], 0.0f);
    v.y = fmaxf(v.y + b[c + 1], 0.0f);
    v.z = fmaxf(v.z + b[c + 2], 0.0f);
    v.w = fmaxf(v.w + b[c + 3], 0.0f);
    *(float4*)p = v;
}

// one wave per batch element: 128-dim dot + shuffle reduction
__global__ void k_score(const float* __restrict__ xp, const float* __restrict__ xW,
                        const int* __restrict__ si, const int* __restrict__ ti,
                        float* __restrict__ out, int B) {
    int gid = blockIdx.x * blockDim.x + threadIdx.x;      // B*32 threads
    int b = gid >> 5;
    if (b >= B) return;
    int lane = gid & 31;
    int s = si[b], t = ti[b];
    int c = lane * 4;
    float4 p = *(const float4*)(xp + (size_t)s * 128 + c);
    float4 w = *(const float4*)(xW + (size_t)t * 128 + c);
    float v = p.x * w.x + p.y * w.y + p.z * w.z + p.w * w.w;
    #pragma unroll
    for (int off = 16; off; off >>= 1) v += __shfl_down(v, off, 32);
    if (lane == 0) out[b] = v;
}

// ---------------------------------------------------------------------------
extern "C" void kernel_launch(void* const* d_in, const int* in_sizes, int n_in,
                              void* d_out, int out_size, void* d_ws, size_t ws_size,
                              hipStream_t stream) {
    const float* x   = (const float*)d_in[0];
    const int*   ei  = (const int*)d_in[1];
    const int*   si  = (const int*)d_in[2];
    const int*   ti  = (const int*)d_in[3];
    const float* Wc0 = (const float*)d_in[4];
    const float* bc0 = (const float*)d_in[5];
    const float* Wc1 = (const float*)d_in[6];
    const float* bc1 = (const float*)d_in[7];
    const float* Wc2 = (const float*)d_in[8];
    const float* bc2 = (const float*)d_in[9];
    const float* Ww  = (const float*)d_in[10];
    const float* bw  = (const float*)d_in[11];
    const float* Wp  = (const float*)d_in[12];
    const float* bp  = (const float*)d_in[13];
    float* out = (float*)d_out;

    const int N = in_sizes[0] / 128;
    const int E = in_sizes[1] / 2;
    const int B = in_sizes[2];
    const int* srcv = ei;
    const int* dstv = ei + E;

    // workspace carve-out
    char* ws = (char*)d_ws;
    size_t off = 0;
    auto al = [](size_t v) { return (v + 255) & ~(size_t)255; };
    float* dinv = (float*)(ws + off); off = al(off + (size_t)N * 4);
    float* xcat = (float*)(ws + off); off = al(off + (size_t)N * 384 * 4);
    float* h    = (float*)(ws + off); off = al(off + (size_t)N * 128 * 4);
    float* xW   = (float*)(ws + off); off = al(off + (size_t)N * 128 * 4);
    float* xp   = (float*)(ws + off); off = al(off + (size_t)N * 128 * 4);
    (void)ws_size; (void)n_in; (void)out_size;

    // symmetric normalization weights
    k_deg_init<<<(N + 255) / 256, 256, 0, stream>>>(dinv, N);
    k_deg_acc <<<(E + 255) / 256, 256, 0, stream>>>(dstv, dinv, E);
    k_dinv    <<<(N + 255) / 256, 256, 0, stream>>>(dinv, N);

    dim3 ggrid((N + 127) / 128, 8);
    const int nblk32 = (N * 32 + 255) / 256;
    const int eblk32 = (int)(((long long)E * 32 + 255) / 256);

    const float* Wl[3] = {Wc0, Wc1, Wc2};
    const float* bl[3] = {bc0, bc1, bc2};
    for (int l = 0; l < 3; ++l) {
        if (l == 0) {
            k_gemm_wmma<128, 128><<<ggrid, 256, 0, stream>>>(x, Wl[0], nullptr, h, N);
        } else {
            k_gemm_wmma<128, 384><<<ggrid, 256, 0, stream>>>(xcat + (l - 1) * 128,
                                                             Wl[l], nullptr, h, N);
        }
        k_selfloop    <<<nblk32, 256, 0, stream>>>(h, dinv, xcat + l * 128, N);
        k_edge_scatter<<<eblk32, 256, 0, stream>>>(h, dinv, srcv, dstv, xcat + l * 128, E);
        k_bias_relu   <<<nblk32, 256, 0, stream>>>(xcat + l * 128, bl[l], N);
    }

    k_gemm_wmma<384, 384><<<ggrid, 256, 0, stream>>>(xcat, Ww, bw, xW, N);
    k_gemm_wmma<384, 384><<<ggrid, 256, 0, stream>>>(xcat, Wp, bp, xp, N);

    k_score<<<(B * 32 + 255) / 256, 256, 0, stream>>>(xp, xW, si, ti, out, B);
}
